// Net_8589934592010
// MI455X (gfx1250) — matrix operand order repaired and checked by reading, not compiled, not run on hardware
//
#include <hip/hip_runtime.h>
#include <hip/hip_bf16.h>

// MI455X / gfx1250 implementation.
// msg_e = sum_h e_h[e,h] * (out[src_e] @ M_h): the 2.4GB W_e tensor is never
// materialized; the 1MB f16 M_h family stays L2-resident and is streamed into
// LDS with GLOBAL_LOAD_ASYNC_TO_LDS_B128 (ASYNCcnt) when available. All heavy
// math (edge einsum + GRU gates) runs on v_wmma_f32_16x16x32_f16.

#define NN   75000
#define EE   150000
#define PP   50000
#define DIM  64
#define HEDGE 128

typedef __attribute__((ext_vector_type(16))) _Float16 v16h;
typedef __attribute__((ext_vector_type(8)))  _Float16 v8h;
typedef __attribute__((ext_vector_type(8)))  float    v8f;
typedef __attribute__((ext_vector_type(4)))  int      v4i;

#if __has_builtin(__builtin_amdgcn_global_load_async_to_lds_b128) && \
    __has_builtin(__builtin_amdgcn_s_wait_asynccnt)
#define USE_ASYNC_LDS 1
// builtin signature (from hipcc diagnostic): (v4i addrspace(1)*, v4i addrspace(3)*, Ii, Ii)
#define AS1V(p) ((__attribute__((address_space(1))) v4i*)(void*)(p))
#define AS3V(p) ((__attribute__((address_space(3))) v4i*)(void*)(p))
#define ASYNC_B128(gp, lp) \
  __builtin_amdgcn_global_load_async_to_lds_b128(AS1V(gp), AS3V(lp), 0, 0)
#else
#define USE_ASYNC_LDS 0
#endif

__device__ __forceinline__ v16h mk16(v8h lo, v8h hi) {
  return __builtin_shufflevector(lo, hi, 0,1,2,3,4,5,6,7,8,9,10,11,12,13,14,15);
}

// ---------------------------------------------------------------- lin0
__global__ __launch_bounds__(256)
void k_lin0(const float* __restrict__ x, const float* __restrict__ w,
            const float* __restrict__ b, float* __restrict__ h0,
            _Float16* __restrict__ outh) {
  int idx = blockIdx.x * blockDim.x + threadIdx.x;
  if (idx >= NN * DIM) return;
  int n = idx / DIM, d = idx % DIM;
  const float* xr = x + (size_t)n * 13;
  const float* wr = w + (size_t)d * 13;
  float acc = b[d];
#pragma unroll
  for (int k = 0; k < 13; ++k) acc += xr[k] * wr[k];
  acc = fmaxf(acc, 0.0f);
  h0[idx] = acc;
  outh[idx] = (_Float16)acc;
}

// ---------------------------------------------------------------- zero / deg / cvt
__global__ void k_zero(float* __restrict__ p, int n) {
  int idx = blockIdx.x * blockDim.x + threadIdx.x;
  if (idx < n) p[idx] = 0.0f;
}

__global__ void k_deg(const int* __restrict__ dst, float* __restrict__ deg) {
  int e = blockIdx.x * blockDim.x + threadIdx.x;
  if (e < EE) atomicAdd(&deg[dst[e]], 1.0f);
}

__global__ void k_h16(const float* __restrict__ s, _Float16* __restrict__ d, int n) {
  int idx = blockIdx.x * blockDim.x + threadIdx.x;
  if (idx < n) d[idx] = (_Float16)s[idx];
}

// ---------------------------------------------------------------- edge MLP stage 1
__global__ __launch_bounds__(256)
void k_eh(const float* __restrict__ ea, const float* __restrict__ w,
          const float* __restrict__ b, _Float16* __restrict__ eh) {
  int idx = blockIdx.x * blockDim.x + threadIdx.x;
  if (idx >= EE * HEDGE) return;
  int e = idx / HEDGE, hd = idx % HEDGE;
  const float* ar = ea + (size_t)e * 4;
  const float* wr = w + (size_t)hd * 4;
  float acc = b[hd] + ar[0]*wr[0] + ar[1]*wr[1] + ar[2]*wr[2] + ar[3]*wr[3];
  eh[idx] = (_Float16)fmaxf(acc, 0.0f);
}

// ------------------------------------------- nn2_w -> Bt[h][o][i] f16 (+bias plane)
__global__ __launch_bounds__(256)
void k_bt(const float* __restrict__ nn2w, const float* __restrict__ nn2b,
          _Float16* __restrict__ Bt) {
  int idx = blockIdx.x * blockDim.x + threadIdx.x;
  if (idx >= 129 * 4096) return;
  int h = idx / 4096;
  int io = idx % 4096;
  int o = io / 64, i = io % 64;
  float v = (h < HEDGE) ? nn2w[(size_t)(i * 64 + o) * HEDGE + h] : nn2b[i * 64 + o];
  Bt[idx] = (_Float16)v;
}

// ---------------------------------------------------------------- fused msg + scatter
__global__ __launch_bounds__(256)
void k_msg(const _Float16* __restrict__ outh, const _Float16* __restrict__ eh,
           const _Float16* __restrict__ Bt, const int* __restrict__ src,
           const int* __restrict__ dst, float* __restrict__ agg) {
  __shared__ __align__(32) _Float16 ldsB[2][DIM * DIM];      // 16 KB double buffer
  __shared__ __align__(16) _Float16 ldsE[8][16 * HEDGE];     // 32 KB e_h staging

  const int tid  = threadIdx.x;
  const int wave = tid >> 5;
  const int lane = tid & 31;
  const int row  = lane & 15;
  const int hi   = lane >> 4;
  const int tileBase = blockIdx.x * 128 + wave * 16;

  // stage this wave's 16 e_h rows (16 x 128 halfs = 256 uint4) into LDS
  {
    const uint4* sp = (const uint4*)eh;
    uint4* dp = (uint4*)(&ldsE[wave][0]);
    for (int i = lane; i < 256; i += 32) {
      int e = tileBase + (i >> 4);
      uint4 v{};
      if (e < EE) v = sp[(size_t)e * 16 + (i & 15)];
      dp[i] = v;
    }
  }

  // raw A fragments (contiguous k order, same in B: lane row=l&15, halves0..15 -> K=16*hi+t)
  int e0 = tileBase + row;
  int srcRow = (e0 < EE) ? src[e0] : 0;
  const _Float16* arow = outh + (size_t)srcRow * DIM;
  const int koff = 8 * hi;
  v16h afrag[2];
  afrag[0] = *(const v16h*)(arow + 16 * hi);
  afrag[1] = *(const v16h*)(arow + 32 + 16 * hi);

  const uint4* bsrc = (const uint4*)Bt;   // 512 uint4 per 8KB M_h chunk
#if USE_ASYNC_LDS
  {
    uint4* l = (uint4*)(&ldsB[0][0]) + tid * 2;
    ASYNC_B128(bsrc + (size_t)tid * 2,     l);
    ASYNC_B128(bsrc + (size_t)tid * 2 + 1, l + 1);
    __builtin_amdgcn_s_wait_asynccnt(0);
  }
  __syncthreads();
#else
  {
    uint4 c0 = bsrc[(size_t)tid * 2 + 0];
    uint4 c1 = bsrc[(size_t)tid * 2 + 1];
    uint4* bl = (uint4*)(&ldsB[0][0]);
    bl[tid * 2 + 0] = c0;
    bl[tid * 2 + 1] = c1;
  }
  __syncthreads();
#endif

  v8f acc[4];
#pragma unroll
  for (int i = 0; i < 4; ++i) acc[i] = (v8f)0.0f;

  for (int h = 0; h <= HEDGE; ++h) {   // h==HEDGE is the nn2_b term (scale 1.0)
    const int cur = h & 1;
#if USE_ASYNC_LDS
    if (h < HEDGE) {                    // buf cur^1 free: all waves past prior barrier
      const uint4* g = bsrc + (size_t)(h + 1) * 512 + tid * 2;
      uint4* l = (uint4*)(&ldsB[cur ^ 1][0]) + tid * 2;
      ASYNC_B128(g,     l);
      ASYNC_B128(g + 1, l + 1);
      if (h + 2 <= HEDGE)
        __builtin_prefetch((const void*)(bsrc + (size_t)(h + 2) * 512 + tid * 2), 0, 1);
    }
#else
    uint4 n0{}, n1{};
    if (h < HEDGE) {
      n0 = bsrc[(size_t)(h + 1) * 512 + tid * 2 + 0];
      n1 = bsrc[(size_t)(h + 1) * 512 + tid * 2 + 1];
      if (h + 2 <= HEDGE)
        __builtin_prefetch((const void*)(bsrc + (size_t)(h + 2) * 512 + tid * 2), 0, 1);
    }
#endif
    _Float16 s = (h < HEDGE) ? ldsE[wave][row * HEDGE + h] : (_Float16)1.0f;

#pragma unroll
    for (int kk = 0; kk < 2; ++kk) {
      v16h aS;
#pragma unroll
      for (int q = 0; q < 16; ++q) aS[q] = afrag[kk][q] * s;  // v_pk_mul_f16
#pragma unroll
      for (int ot = 0; ot < 4; ++ot) {
        const _Float16* bp = &ldsB[cur][(ot * 16 + row) * DIM + kk * 32 + 16 * hi];
        v16h bF = *(const v16h*)bp;
        acc[ot] = __builtin_amdgcn_wmma_f32_16x16x32_f16(
            false, aS, false, bF, (short)0, acc[ot], false, false);
      }
    }
#if USE_ASYNC_LDS
    if (h < HEDGE) {
      __builtin_amdgcn_s_wait_asynccnt(0);
      __syncthreads();
    }
#else
    __syncthreads();
    if (h < HEDGE) {
      uint4* bl = (uint4*)(&ldsB[cur ^ 1][0]);
      bl[tid * 2 + 0] = n0;
      bl[tid * 2 + 1] = n1;
      __syncthreads();
    }
#endif
  }

  // scatter: lane(row,hi), vgpr r holds msg[edge tileBase+r+8*hi, ot*16+row]
  int ebase = tileBase + 8 * hi;
  int dsts[8];
#pragma unroll
  for (int r = 0; r < 8; ++r) {
    int e = ebase + r;
    dsts[r] = (e < EE) ? dst[e] : -1;
  }
#pragma unroll
  for (int ot = 0; ot < 4; ++ot)
#pragma unroll
    for (int r = 0; r < 8; ++r)
      if (dsts[r] >= 0)
        atomicAdd(&agg[(size_t)dsts[r] * DIM + ot * 16 + row], acc[ot][r]);
}

// ---------------------------------------------------------------- mean + bias + relu
__global__ __launch_bounds__(256)
void k_norm(float* __restrict__ agg, _Float16* __restrict__ m16,
            const float* __restrict__ deg, const float* __restrict__ convb) {
  int idx = blockIdx.x * blockDim.x + threadIdx.x;
  if (idx >= NN * DIM) return;
  int n = idx / DIM, d = idx % DIM;
  float dg = fmaxf(deg[n], 1.0f);
  float v = fmaxf(agg[idx] / dg + convb[d], 0.0f);
  agg[idx] = v;
  m16[idx] = (_Float16)v;
}

// ---------------------------------------------------------------- fused WMMA GRU
// gi = m@W_ih.T, gh = h@W_hh.T as 16x16x32 WMMA with both weight mats in LDS.
// Gate columns d, 64+d, 128+d land in the SAME lane of n-tiles ot, ot+4, ot+8,
// so sigmoid/tanh gate math runs directly on accumulators; h updates in place.
__global__ __launch_bounds__(256)
void k_gruw(const _Float16* __restrict__ m16, const _Float16* __restrict__ h16,
            const _Float16* __restrict__ wih16, const _Float16* __restrict__ whh16,
            const float* __restrict__ bih, const float* __restrict__ bhh,
            float* __restrict__ h, _Float16* __restrict__ outh) {
  __shared__ __align__(32) _Float16 sWI[192 * DIM];   // 24 KB
  __shared__ __align__(32) _Float16 sWH[192 * DIM];   // 24 KB
  const int tid  = threadIdx.x;
  const int wave = tid >> 5;
  const int lane = tid & 31;
  const int row  = lane & 15;
  const int hi   = lane >> 4;

  {
    const uint4* s1 = (const uint4*)wih16;
    const uint4* s2 = (const uint4*)whh16;
    uint4* d1 = (uint4*)sWI;
    uint4* d2 = (uint4*)sWH;
    for (int i = tid; i < 1536; i += 256) { d1[i] = s1[i]; d2[i] = s2[i]; }
  }

  const int base = blockIdx.x * 128 + wave * 16;
  int nr = base + row;
  int arow = (nr < NN) ? nr : (NN - 1);
  const _Float16* mr = m16 + (size_t)arow * DIM;
  const _Float16* hr = h16 + (size_t)arow * DIM;
  const int koff = 8 * hi;
  v16h am[2], ah[2];
  am[0] = *(const v16h*)(mr + 16 * hi);
  am[1] = *(const v16h*)(mr + 32 + 16 * hi);
  ah[0] = *(const v16h*)(hr + 16 * hi);
  ah[1] = *(const v16h*)(hr + 32 + 16 * hi);
  __syncthreads();

  v8f aI[12], aH[12];
#pragma unroll
  for (int i = 0; i < 12; ++i) { aI[i] = (v8f)0.0f; aH[i] = (v8f)0.0f; }

#pragma unroll
  for (int kk = 0; kk < 2; ++kk) {
#pragma unroll
    for (int ot = 0; ot < 12; ++ot) {
      const int bo = (ot * 16 + row) * DIM + kk * 32 + 16 * hi;
      v16h b1 = *(const v16h*)(&sWI[bo]);
      aI[ot] = __builtin_amdgcn_wmma_f32_16x16x32_f16(
          false, am[kk], false, b1, (short)0, aI[ot], false, false);
      v16h b2 = *(const v16h*)(&sWH[bo]);
      aH[ot] = __builtin_amdgcn_wmma_f32_16x16x32_f16(
          false, ah[kk], false, b2, (short)0, aH[ot], false, false);
    }
  }

#pragma unroll
  for (int ot = 0; ot < 4; ++ot) {
    const int d = ot * 16 + row;
    float bir = bih[d], biz = bih[64 + d], bin = bih[128 + d];
    float bhr = bhh[d], bhz = bhh[64 + d], bhn = bhh[128 + d];
#pragma unroll
    for (int r = 0; r < 8; ++r) {
      int n = base + r + 8 * hi;
      if (n < NN) {
        float ir  = aI[ot][r]     + bir;
        float iz  = aI[ot + 4][r] + biz;
        float in_ = aI[ot + 8][r] + bin;
        float hrv = aH[ot][r]     + bhr;
        float hz  = aH[ot + 4][r] + bhz;
        float hn  = aH[ot + 8][r] + bhn;
        float rr = 1.0f / (1.0f + __expf(-(ir + hrv)));
        float zz = 1.0f / (1.0f + __expf(-(iz + hz)));
        float ng = tanhf(in_ + rr * hn);
        float hp = h[(size_t)n * DIM + d];
        float hnew = (1.0f - zz) * ng + zz * hp;
        h[(size_t)n * DIM + d] = hnew;
        outh[(size_t)n * DIM + d] = (_Float16)hnew;
      }
    }
  }
}

// ---------------------------------------------------------------- head fold
__global__ void k_headfold(const float* __restrict__ l1w, const float* __restrict__ l1b,
                           const float* __restrict__ l2w, const float* __restrict__ l2b,
                           float* __restrict__ v) {
  int o = threadIdx.x;  // 64 threads
  float acc = 0.0f;
  for (int d = 0; d < DIM; ++d)
    acc += l2w[d] * (l1w[(size_t)d * 128 + o] + l1w[(size_t)d * 128 + 64 + o]);
  v[o] = acc;
  if (o == 0) {
    float c = l2b[0];
    for (int d = 0; d < DIM; ++d) c += l2w[d] * l1b[d];
    v[64] = c;
  }
}

// ---------------------------------------------------------------- head
__global__ __launch_bounds__(256)
void k_head(const float* __restrict__ hfin, const int* __restrict__ atom0,
            const float* __restrict__ v, float* __restrict__ out) {
  __shared__ float sv[65];
  if (threadIdx.x < 65) sv[threadIdx.x] = v[threadIdx.x];
  __syncthreads();
  int p = blockIdx.x * blockDim.x + threadIdx.x;
  if (p >= PP) return;
  const float* row = hfin + (size_t)atom0[p] * DIM;
  float acc = sv[64];
#pragma unroll 8
  for (int k = 0; k < DIM; ++k) acc += row[k] * sv[k];
  out[p] = acc;
}

// ---------------------------------------------------------------- launch
extern "C" void kernel_launch(void* const* d_in, const int* in_sizes, int n_in,
                              void* d_out, int out_size, void* d_ws, size_t ws_size,
                              hipStream_t stream) {
  (void)in_sizes; (void)n_in; (void)out_size; (void)ws_size;
  const float* x         = (const float*)d_in[0];
  const float* edge_attr = (const float*)d_in[1];
  const int*   ei        = (const int*)d_in[2];
  const int*   ti        = (const int*)d_in[3];
  const float* lin0w     = (const float*)d_in[4];
  const float* lin0b     = (const float*)d_in[5];
  const float* nn1w      = (const float*)d_in[6];
  const float* nn1b      = (const float*)d_in[7];
  const float* nn2w      = (const float*)d_in[8];
  const float* nn2b      = (const float*)d_in[9];
  const float* convb     = (const float*)d_in[10];
  const float* wih       = (const float*)d_in[11];
  const float* whh       = (const float*)d_in[12];
  const float* bih       = (const float*)d_in[13];
  const float* bhh       = (const float*)d_in[14];
  const float* l1w       = (const float*)d_in[15];
  const float* l1b       = (const float*)d_in[16];
  const float* l2w       = (const float*)d_in[17];
  const float* l2b       = (const float*)d_in[18];
  const int* srcIdx = ei;        // edge_index[0]
  const int* dstIdx = ei + EE;   // edge_index[1]
  const int* atom0  = ti;        // target_indices[0]

  // workspace carve (~98 MB)
  char* ws = (char*)d_ws;
  auto take = [&](size_t bytes) -> char* {
    char* p = ws;
    ws += (bytes + 255) & ~(size_t)255;
    return p;
  };
  float*    hA    = (float*)take((size_t)NN * DIM * 4);
  _Float16* outh  = (_Float16*)take((size_t)NN * DIM * 2);
  _Float16* m16   = (_Float16*)take((size_t)NN * DIM * 2);
  _Float16* eh16  = (_Float16*)take((size_t)EE * HEDGE * 2);
  _Float16* Bt    = (_Float16*)take((size_t)129 * 4096 * 2);
  float*    agg   = (float*)take((size_t)NN * DIM * 4);
  float*    deg   = (float*)take((size_t)NN * 4);
  _Float16* wih16 = (_Float16*)take((size_t)192 * DIM * 2);
  _Float16* whh16 = (_Float16*)take((size_t)192 * DIM * 2);
  float*    vh    = (float*)take(65 * 4);

  auto cdiv = [](long a, long b) { return (int)((a + b - 1) / b); };
  dim3 blk(256);

  k_lin0<<<cdiv((long)NN * DIM, 256), blk, 0, stream>>>(x, lin0w, lin0b, hA, outh);
  k_zero<<<cdiv(NN, 256), blk, 0, stream>>>(deg, NN);
  k_deg<<<cdiv(EE, 256), blk, 0, stream>>>(dstIdx, deg);
  k_eh<<<cdiv((long)EE * HEDGE, 256), blk, 0, stream>>>(edge_attr, nn1w, nn1b, eh16);
  k_bt<<<cdiv(129L * 4096, 256), blk, 0, stream>>>(nn2w, nn2b, Bt);
  k_h16<<<cdiv(192L * DIM, 256), blk, 0, stream>>>(wih, wih16, 192 * DIM);
  k_h16<<<cdiv(192L * DIM, 256), blk, 0, stream>>>(whh, whh16, 192 * DIM);

  for (int s = 0; s < 3; ++s) {
    k_zero<<<cdiv((long)NN * DIM, 256), blk, 0, stream>>>(agg, NN * DIM);
    k_msg<<<cdiv(EE, 128), blk, 0, stream>>>(outh, eh16, Bt, srcIdx, dstIdx, agg);
    k_norm<<<cdiv((long)NN * DIM, 256), blk, 0, stream>>>(agg, m16, deg, convb);
    k_gruw<<<cdiv(NN, 128), blk, 0, stream>>>(m16, outh, wih16, whh16, bih, bhh, hA, outh);
  }

  k_headfold<<<1, 64, 0, stream>>>(l1w, l1b, l2w, l2b, vh);
  k_head<<<cdiv(PP, 256), blk, 0, stream>>>(hA, atom0, vh, (float*)d_out);
}
